// GATEncoder_89111981457475
// MI455X (gfx1250) — matrix operand order repaired and checked by reading, not compiled, not run on hardware
//
#include <hip/hip_runtime.h>
#include <hip/hip_bf16.h>
#include <math.h>

// ---------------------------------------------------------------------------
// GAT encoder for MI455X (gfx1250): bf16 WMMA GEMMs + L2-resident edge phases
// ---------------------------------------------------------------------------

#define IN_C   617
#define KPAD   640          // 617 padded to multiple of 32
#define HID    64
#define HEADS  4
#define F1     256          // HEADS*HID
#define OUT_C  16
#define NEG_SLOPE 0.2f
#define EPS_SM 1e-16f
#define TPB 256

typedef __attribute__((ext_vector_type(16))) __bf16 bf16x16;
typedef __attribute__((ext_vector_type(8)))  __bf16 bf16x8;
typedef __attribute__((ext_vector_type(8)))  float  f32x8;

__device__ __forceinline__ unsigned short f2bf(float f) {
    unsigned u = __float_as_uint(f);
    unsigned r = u + 0x7FFFu + ((u >> 16) & 1u);   // round-to-nearest-even
    return (unsigned short)(r >> 16);
}

__device__ __forceinline__ bf16x16 cat16(bf16x8 lo, bf16x8 hi) {
    return __builtin_shufflevector(lo, hi, 0,1,2,3,4,5,6,7,8,9,10,11,12,13,14,15);
}

__device__ __forceinline__ f32x8 wmma_bf16(bf16x16 a, bf16x16 b, f32x8 c) {
    // emits v_wmma_f32_16x16x32_bf16
    return __builtin_amdgcn_wmma_f32_16x16x32_bf16(false, a, false, b, (short)0, c,
                                                   false, false);
}

__device__ __forceinline__ void atomicMaxFloat(float* addr, float val) {
    // sign-aware monotonic mapping of IEEE-754 to int/uint ordering
    if (val >= 0.f) atomicMax((int*)addr, __float_as_int(val));
    else            atomicMin((unsigned int*)addr, __float_as_uint(val));
}

__device__ __forceinline__ float lrelu(float v) {
    return v > 0.f ? v : NEG_SLOPE * v;
}

// ---------------------------------------------------------------------------
// Weight conversion
// ---------------------------------------------------------------------------

// W1 [617][256] f32 -> COLUMN-MAJOR bf16 [256 cols][640 k], zero padded K
__global__ void conv_w1_kernel(const float* __restrict__ W1,
                               unsigned short* __restrict__ w1cm) {
    int t = blockIdx.x * blockDim.x + threadIdx.x;
    if (t >= F1 * KPAD) return;
    int col = t / KPAD, k = t - col * KPAD;
    float v = (k < IN_C) ? W1[k * F1 + col] : 0.f;
    w1cm[t] = f2bf(v);
}

// W2 [256][16] f32 -> column-major bf16 [16][256]
__global__ void conv_w2_kernel(const float* __restrict__ W2,
                               unsigned short* __restrict__ w2cm) {
    int t = blockIdx.x * blockDim.x + threadIdx.x;
    if (t >= OUT_C * F1) return;
    int col = t >> 8, k = t & 255;
    w2cm[t] = f2bf(W2[k * OUT_C + col]);
}

// ---------------------------------------------------------------------------
// GEMM1: h1[N][256] = x[N][617] @ W1  (bf16 WMMA, f32 accumulate)
// Block: 256 threads = 8 waves in 2x4; tile 64x64, BK=32, double-buffered LDS.
// Staging is fully branchless: clamped addresses + multiply-by-mask zeroing.
// Store epilogue: uniform block-level bounds test -> unguarded fast path.
// ---------------------------------------------------------------------------
#define BM 64
#define BN 64
#define BK 32
#define NSTEP (KPAD / BK)   // 20
#define LDS_S 40            // ushort stride (80B rows: 16B-aligned, spreads banks)

__global__ void gemm1_kernel(const float* __restrict__ x,
                             const unsigned short* __restrict__ w1cm,
                             float* __restrict__ h1, int Nn) {
    __shared__ __align__(16) unsigned short As[2][BM][LDS_S]; // row-major  [m][k]
    __shared__ __align__(16) unsigned short Bs[2][BN][LDS_S]; // col-major  [n][k]

    const int tid  = threadIdx.x;
    const int lane = tid & 31;
    const int wave = tid >> 5;
    const int wm   = wave >> 2;      // 0..1  -> 32-row slab
    const int wn   = wave & 3;       // 0..3  -> 16-col slab
    const int row0 = blockIdx.x * BM;
    const int col0 = blockIdx.y * BN;

    // staging coords: thread t fills A[ar][ac..ac+7] and B[ar][ac..ac+7]
    const int ar  = tid >> 2;            // 0..63
    const int ac  = (tid & 3) * 8;       // 0,8,16,24
    const int grow = min(row0 + ar, Nn - 1);           // clamped: always valid
    const float* __restrict__ xrow = x + (long)grow * IN_C;
    const unsigned short* __restrict__ wcol = w1cm + (long)(col0 + ar) * KPAD;

    // fragment coords (ISA 7.12.2 16-bit A 16x32 / B 32x16 layouts)
    const int l15 = lane & 15;
    const int klo = (lane >> 4) * 8;     // A frag k-offset: 0 or 8
    const int kb  = (lane >> 4) * 8;     // B frag k-offset: 0 or 8

    f32x8 acc0 = {}; f32x8 acc1 = {};

    auto stage = [&](int kk, int buf) {
        union { unsigned short us[8]; uint4 v; } tmp;
        #pragma unroll
        for (int j = 0; j < 8; ++j) {
            int gc  = kk + ac + j;
            int gcl = gc < IN_C ? gc : IN_C - 1;       // v_cndmask, valid address
            float lv = xrow[gcl];                      // unconditional load
            float mk = gc < IN_C ? 1.f : 0.f;          // v_cndmask
            tmp.us[j] = f2bf(lv * mk);                 // load cannot be re-predicated
        }
        *reinterpret_cast<uint4*>(&As[buf][ar][ac]) = tmp.v;
        *reinterpret_cast<uint4*>(&Bs[buf][ar][ac]) =
            *reinterpret_cast<const uint4*>(wcol + kk + ac);
        // speculative prefetch of the K tile after next (global_prefetch_b8)
        __builtin_prefetch(xrow + kk + 2 * BK + ac, 0, 1);
    };

    stage(0, 0);
    for (int s = 0; s < NSTEP; ++s) {
        __syncthreads();                       // staged buf (s&1) visible
        if (s + 1 < NSTEP) stage((s + 1) * BK, (s + 1) & 1);
        const int buf = s & 1;
        const int arow0 = wm * 32 + l15;
        bf16x16 a0 = cat16(*reinterpret_cast<const bf16x8*>(&As[buf][arow0][klo]),
                           *reinterpret_cast<const bf16x8*>(&As[buf][arow0][klo + 16]));
        bf16x16 a1 = cat16(*reinterpret_cast<const bf16x8*>(&As[buf][arow0 + 16][klo]),
                           *reinterpret_cast<const bf16x8*>(&As[buf][arow0 + 16][klo + 16]));
        const int bcol = wn * 16 + l15;
        bf16x16 bfrag = cat16(*reinterpret_cast<const bf16x8*>(&Bs[buf][bcol][kb]),
                              *reinterpret_cast<const bf16x8*>(&Bs[buf][bcol][kb + 16]));
        acc0 = wmma_bf16(a0, bfrag, acc0);
        acc1 = wmma_bf16(a1, bfrag, acc1);
    }

    // ---- store D (VGPR i: lane<16 -> M=i, lane>=16 -> M=8+i; N=lane&15) ----
    const int ccol  = col0 + wn * 16 + l15;
    const int rbase = row0 + wm * 32 + (lane >> 4) * 8;
    float* hp = h1 + (long)rbase * F1 + ccol;
    if (row0 + BM <= Nn) {            // uniform: all rows valid (781/782 blocks)
        #pragma unroll
        for (int i = 0; i < 8; ++i) {
            hp[(long)i * F1]        = acc0[i];
            hp[(long)(i + 16) * F1] = acc1[i];
        }
    } else {                          // boundary block only
        #pragma unroll
        for (int i = 0; i < 8; ++i) {
            int r = rbase + i;
            if (r < Nn)      h1[(long)r * F1 + ccol]        = acc0[i];
            if (r + 16 < Nn) h1[(long)(r + 16) * F1 + ccol] = acc1[i];
        }
    }
}

// ---------------------------------------------------------------------------
// GEMM2: h2[N][16] = g_bf16[N][256] @ W2 (bf16 WMMA). 8 waves = 128 rows/block.
// Fragments loaded directly from global (A lane loads are 16B aligned).
// ---------------------------------------------------------------------------
__global__ void gemm2_kernel(const unsigned short* __restrict__ g,
                             const unsigned short* __restrict__ w2cm,
                             float* __restrict__ h2, int Nn) {
    const int tid  = threadIdx.x;
    const int lane = tid & 31;
    const int wave = tid >> 5;
    const int row0 = blockIdx.x * 128 + wave * 16;
    const int l15  = lane & 15;
    const int hi   = lane >> 4;

    const int  arow  = row0 + l15;
    const long abase = (long)((arow < Nn) ? arow : 0) * F1;

    f32x8 acc = {};
    #pragma unroll
    for (int kk = 0; kk < F1; kk += 32) {
        const int ka = kk + hi * 8;
        bf16x16 a = cat16(*reinterpret_cast<const bf16x8*>(g + abase + ka),
                          *reinterpret_cast<const bf16x8*>(g + abase + ka + 16));
        const int kbb = kk + hi * 8;
        const unsigned short* bp = w2cm + l15 * F1 + kbb;
        bf16x16 b = cat16(*reinterpret_cast<const bf16x8*>(bp),
                          *reinterpret_cast<const bf16x8*>(bp + 16));
        acc = wmma_bf16(a, b, acc);
    }
    float* hp = h2 + (long)(row0 + hi * 8) * OUT_C + l15;
    if (row0 + 16 <= Nn) {            // uniform per wave: fast path
        #pragma unroll
        for (int i = 0; i < 8; ++i) hp[i * OUT_C] = acc[i];
    } else {
        #pragma unroll
        for (int i = 0; i < 8; ++i) {
            int r = row0 + hi * 8 + i;
            if (r < Nn) h2[(long)r * OUT_C + l15] = acc[i];
        }
    }
}

// ---------------------------------------------------------------------------
// Per-node attention coefficients: alpha_s[n,h] = <h[n,h,:], a_src[h,:]>
// ---------------------------------------------------------------------------
__global__ void node_alpha_kernel(const float* __restrict__ h,
                                  const float* __restrict__ a_s,
                                  const float* __restrict__ a_d,
                                  float* __restrict__ as_out,
                                  float* __restrict__ ad_out,
                                  int Nn, int H, int C) {
    int t = blockIdx.x * blockDim.x + threadIdx.x;
    if (t >= Nn * H) return;
    int n = t / H, hd = t - n * H;
    const float* hp = h + (long)n * H * C + hd * C;
    const float* asv = a_s + hd * C;
    const float* adv = a_d + hd * C;
    float s = 0.f, d = 0.f;
    for (int c = 0; c < C; ++c) { float v = hp[c]; s += v * asv[c]; d += v * adv[c]; }
    as_out[t] = s; ad_out[t] = d;
}

// ---------------------------------------------------------------------------
// Softmax init / edge passes (edges E + N self-loops)
// ---------------------------------------------------------------------------
__device__ __forceinline__ void edge_sd(const long long* ei, int e, int E,
                                        long long& s, long long& d) {
    if (e < E) { s = ei[e]; d = ei[(long long)E + e]; }
    else       { s = d = (long long)(e - E); }
}

__global__ void init_l1_kernel(float* __restrict__ hagg, float* __restrict__ m1,
                               float* __restrict__ den1, int Nn) {
    int t = blockIdx.x * blockDim.x + threadIdx.x;
    if (t < Nn * F1) hagg[t] = 0.f;
    if (t < Nn * HEADS) { m1[t] = __int_as_float(0xFF800000); den1[t] = 0.f; }
}

__global__ void init_l2_kernel(float* __restrict__ m2, float* __restrict__ den2,
                               float* __restrict__ out, const float* __restrict__ b2,
                               int Nn) {
    int t = blockIdx.x * blockDim.x + threadIdx.x;
    if (t < Nn * OUT_C) out[t] = b2[t & (OUT_C - 1)];
    if (t < Nn) { m2[t] = __int_as_float(0xFF800000); den2[t] = 0.f; }
}

__global__ void edge_max_kernel(const long long* __restrict__ ei,
                                const float* __restrict__ as,
                                const float* __restrict__ ad,
                                float* __restrict__ m, int Nn, int E, int H) {
    int t = blockIdx.x * blockDim.x + threadIdx.x;
    int tot = (E + Nn) * H;
    if (t >= tot) return;
    int e = t / H, hd = t - e * H;
    long long s, d; edge_sd(ei, e, E, s, d);
    float v = lrelu(as[s * H + hd] + ad[d * H + hd]);
    atomicMaxFloat(&m[d * H + hd], v);
}

__global__ void edge_sum_kernel(const long long* __restrict__ ei,
                                const float* __restrict__ as,
                                const float* __restrict__ ad,
                                const float* __restrict__ m,
                                float* __restrict__ den, int Nn, int E, int H) {
    int t = blockIdx.x * blockDim.x + threadIdx.x;
    int tot = (E + Nn) * H;
    if (t >= tot) return;
    int e = t / H, hd = t - e * H;
    long long s, d; edge_sd(ei, e, E, s, d);
    float v = lrelu(as[s * H + hd] + ad[d * H + hd]);
    atomicAdd(&den[d * H + hd], __expf(v - m[d * H + hd]));
}

// Layer-1 aggregation: one wave per edge, 8 channels per lane (head = lane>>3)
__global__ void eagg1_kernel(const long long* __restrict__ ei,
                             const float* __restrict__ as,
                             const float* __restrict__ ad,
                             const float* __restrict__ m,
                             const float* __restrict__ den,
                             const float* __restrict__ h1,
                             float* __restrict__ hagg, int Nn, int E) {
    int gw   = (blockIdx.x * blockDim.x + threadIdx.x) >> 5;   // edge id
    int lane = threadIdx.x & 31;
    if (gw >= E + Nn) return;
    long long s, d; edge_sd(ei, gw, E, s, d);
    int hd = lane >> 3;                                        // 8 ch per lane, 64/head
    float v = lrelu(as[s * HEADS + hd] + ad[d * HEADS + hd]);
    float alpha = __expf(v - m[d * HEADS + hd]) / (den[d * HEADS + hd] + EPS_SM);
    const float4* hp = reinterpret_cast<const float4*>(h1 + s * F1) + lane * 2;
    float4 v0 = hp[0], v1 = hp[1];
    float* op = hagg + d * F1 + lane * 8;
    atomicAdd(op + 0, alpha * v0.x); atomicAdd(op + 1, alpha * v0.y);
    atomicAdd(op + 2, alpha * v0.z); atomicAdd(op + 3, alpha * v0.w);
    atomicAdd(op + 4, alpha * v1.x); atomicAdd(op + 5, alpha * v1.y);
    atomicAdd(op + 6, alpha * v1.z); atomicAdd(op + 7, alpha * v1.w);
}

// out = elu(hagg + b1) -> bf16 for GEMM2
__global__ void elu_conv_kernel(const float* __restrict__ hagg,
                                const float* __restrict__ b1,
                                unsigned short* __restrict__ g, int Nn) {
    int t = blockIdx.x * blockDim.x + threadIdx.x;
    if (t >= Nn * F1) return;
    float v = hagg[t] + b1[t & (F1 - 1)];
    v = v > 0.f ? v : expm1f(v);
    g[t] = f2bf(v);
}

// Layer-2 aggregation: thread per (edge, channel); out pre-initialized with b2
__global__ void eagg2_kernel(const long long* __restrict__ ei,
                             const float* __restrict__ as,
                             const float* __restrict__ ad,
                             const float* __restrict__ m,
                             const float* __restrict__ den,
                             const float* __restrict__ h2,
                             float* __restrict__ out, int Nn, int E) {
    int t = blockIdx.x * blockDim.x + threadIdx.x;
    if (t >= (E + Nn) * OUT_C) return;
    int e = t >> 4, c = t & (OUT_C - 1);
    long long s, d; edge_sd(ei, e, E, s, d);
    float v = lrelu(as[s] + ad[d]);
    float alpha = __expf(v - m[d]) / (den[d] + EPS_SM);
    atomicAdd(&out[d * OUT_C + c], alpha * h2[s * OUT_C + c]);
}

// ---------------------------------------------------------------------------
// Host launcher
// ---------------------------------------------------------------------------
extern "C" void kernel_launch(void* const* d_in, const int* in_sizes, int n_in,
                              void* d_out, int out_size, void* d_ws, size_t ws_size,
                              hipStream_t stream) {
    const float*     x       = (const float*)d_in[0];
    const long long* ei      = (const long long*)d_in[1];   // int64 edge_index [2][E]
    const float*     W1      = (const float*)d_in[2];
    const float*     a_src1  = (const float*)d_in[3];
    const float*     a_dst1  = (const float*)d_in[4];
    const float*     b1      = (const float*)d_in[5];
    const float*     W2      = (const float*)d_in[6];
    const float*     a_src2  = (const float*)d_in[7];
    const float*     a_dst2  = (const float*)d_in[8];
    const float*     b2      = (const float*)d_in[9];
    float*           out     = (float*)d_out;

    const int Nn = in_sizes[0] / IN_C;      // 50000
    const int E  = in_sizes[1] / 2;         // 800000
    const int ET = E + Nn;                  // edges incl. self-loops

    // workspace carving (256B aligned)
    char* ws = (char*)d_ws;
    size_t off = 0;
    auto alloc = [&](size_t bytes) -> void* {
        off = (off + 255) & ~(size_t)255;
        void* p = ws + off; off += bytes; return p;
    };
    unsigned short* w1cm = (unsigned short*)alloc((size_t)F1 * KPAD * 2);
    unsigned short* w2cm = (unsigned short*)alloc((size_t)OUT_C * F1 * 2);
    float* h1   = (float*)alloc((size_t)Nn * F1 * 4);
    float* hagg = (float*)alloc((size_t)Nn * F1 * 4);
    unsigned short* gbf = (unsigned short*)alloc((size_t)Nn * F1 * 2);
    float* h2   = (float*)alloc((size_t)Nn * OUT_C * 4);
    float* as1  = (float*)alloc((size_t)Nn * HEADS * 4);
    float* ad1  = (float*)alloc((size_t)Nn * HEADS * 4);
    float* m1   = (float*)alloc((size_t)Nn * HEADS * 4);
    float* den1 = (float*)alloc((size_t)Nn * HEADS * 4);
    float* as2  = (float*)alloc((size_t)Nn * 4);
    float* ad2  = (float*)alloc((size_t)Nn * 4);
    float* m2   = (float*)alloc((size_t)Nn * 4);
    float* den2 = (float*)alloc((size_t)Nn * 4);

    auto blocks = [](long total) { return (int)((total + TPB - 1) / TPB); };

    // weights -> bf16 (W1 column-major, W2 column-major)
    conv_w1_kernel<<<blocks((long)F1 * KPAD), TPB, 0, stream>>>(W1, w1cm);
    conv_w2_kernel<<<blocks((long)OUT_C * F1), TPB, 0, stream>>>(W2, w2cm);

    // layer 1: GEMM + attention + softmax + aggregation
    gemm1_kernel<<<dim3((Nn + BM - 1) / BM, F1 / BN), TPB, 0, stream>>>(x, w1cm, h1, Nn);
    node_alpha_kernel<<<blocks((long)Nn * HEADS), TPB, 0, stream>>>(
        h1, a_src1, a_dst1, as1, ad1, Nn, HEADS, HID);
    init_l1_kernel<<<blocks((long)Nn * F1), TPB, 0, stream>>>(hagg, m1, den1, Nn);
    edge_max_kernel<<<blocks((long)ET * HEADS), TPB, 0, stream>>>(ei, as1, ad1, m1, Nn, E, HEADS);
    edge_sum_kernel<<<blocks((long)ET * HEADS), TPB, 0, stream>>>(ei, as1, ad1, m1, den1, Nn, E, HEADS);
    eagg1_kernel<<<blocks((long)ET * 32), TPB, 0, stream>>>(ei, as1, ad1, m1, den1, h1, hagg, Nn, E);

    // ELU(+b1) -> bf16, layer 2
    elu_conv_kernel<<<blocks((long)Nn * F1), TPB, 0, stream>>>(hagg, b1, gbf, Nn);
    gemm2_kernel<<<(Nn + 127) / 128, TPB, 0, stream>>>(gbf, w2cm, h2, Nn);
    node_alpha_kernel<<<blocks((long)Nn), TPB, 0, stream>>>(
        h2, a_src2, a_dst2, as2, ad2, Nn, 1, OUT_C);
    init_l2_kernel<<<blocks((long)Nn * OUT_C), TPB, 0, stream>>>(m2, den2, out, b2, Nn);
    edge_max_kernel<<<blocks((long)ET), TPB, 0, stream>>>(ei, as2, ad2, m2, Nn, E, 1);
    edge_sum_kernel<<<blocks((long)ET), TPB, 0, stream>>>(ei, as2, ad2, m2, den2, Nn, E, 1);
    eagg2_kernel<<<blocks((long)ET * OUT_C), TPB, 0, stream>>>(ei, as2, ad2, m2, den2, h2, out, Nn, E);
}
